// LlamaAttention_37426345018075
// MI455X (gfx1250) — matrix operand order repaired and checked by reading, not compiled, not run on hardware
//
#include <hip/hip_runtime.h>
#include <hip/hip_bf16.h>

typedef __attribute__((ext_vector_type(16))) __bf16 bf16x16;
typedef __attribute__((ext_vector_type(8)))  __bf16 bf16x8;
typedef __attribute__((ext_vector_type(4)))  __bf16 bf16x4;
typedef __attribute__((ext_vector_type(8)))  float  f32x8;

#define B_   2
#define S_   2048
#define H_   4096
#define NH_  32
#define HD_  128

__device__ __forceinline__ f32x8 vzero8() {
  f32x8 z;
#pragma unroll
  for (int i = 0; i < 8; ++i) z[i] = 0.f;
  return z;
}

__device__ __forceinline__ bf16x16 mk16(bf16x8 lo, bf16x8 hi) {
  bf16x16 r;
#pragma unroll
  for (int i = 0; i < 8; ++i) { r[i] = lo[i]; r[i + 8] = hi[i]; }
  return r;
}

// ---- CDNA5 async global->LDS copy (ASYNCcnt-tracked, bypasses VGPRs) -------
// Generic pointer for an LDS object carries the aperture in the high 32 bits;
// the low 32 bits are the DS byte offset the instruction's VDST expects.
__device__ __forceinline__ void async_copy_b128(const __bf16* g, __bf16* l) {
  unsigned loff = (unsigned)(unsigned long long)(uintptr_t)l;
  unsigned long long ga = (unsigned long long)(uintptr_t)g;
  asm volatile("global_load_async_to_lds_b128 %0, %1, off"
               :: "v"(loff), "v"(ga) : "memory");
}
__device__ __forceinline__ void wait_async_le2() {
  asm volatile("s_wait_asynccnt 2" ::: "memory");
}
__device__ __forceinline__ void wait_async_le0() {
  asm volatile("s_wait_asynccnt 0" ::: "memory");
}

// ---------------------------------------------------------------- conversions
__global__ __launch_bounds__(256) void f32_to_bf16_k(const float* __restrict__ src,
                                                     __bf16* __restrict__ dst, int n) {
  int i = (blockIdx.x * 256 + threadIdx.x) * 4;
  if (i >= n) return;
  float4 f = *(const float4*)(src + i);
  bf16x4 o;
  o[0] = (__bf16)f.x; o[1] = (__bf16)f.y; o[2] = (__bf16)f.z; o[3] = (__bf16)f.w;
  *(bf16x4*)(dst + i) = o;
}

// ---------------------------------------------------------------- RoPE tables
__global__ __launch_bounds__(256) void rope_tables_k(float* __restrict__ cosT,
                                                     float* __restrict__ sinT) {
  int i = blockIdx.x * 256 + threadIdx.x;        // S_*64
  if (i >= S_ * 64) return;
  int s = i >> 6, d = i & 63;
  float freq = __powf(10000.f, -(float)d * (1.f / 64.f));
  float ang = (float)s * freq;
  cosT[i] = cosf(ang);
  sinT[i] = sinf(ang);
}

// in-place RoPE on Q and K, layout [token(4096), H_] with col = nh*128 + d
__global__ __launch_bounds__(256) void rope_apply_k(__bf16* __restrict__ Q,
                                                    __bf16* __restrict__ K,
                                                    const float* __restrict__ cosT,
                                                    const float* __restrict__ sinT) {
  int i = blockIdx.x * 256 + threadIdx.x;        // 4096 tokens * 32 heads * 64 pairs
  if (i >= (B_ * S_) * NH_ * 64) return;
  int d   = i & 63;
  int nh  = (i >> 6) & (NH_ - 1);
  int tok = i >> 11;
  int s   = tok & (S_ - 1);
  float c  = cosT[(s << 6) + d];
  float sn = sinT[(s << 6) + d];
  size_t base = ((size_t)tok << 12) + ((size_t)nh << 7) + d;
  float q1 = (float)Q[base], q2 = (float)Q[base + 64];
  Q[base]      = (__bf16)(q1 * c - q2 * sn);
  Q[base + 64] = (__bf16)(q2 * c + q1 * sn);
  float k1 = (float)K[base], k2 = (float)K[base + 64];
  K[base]      = (__bf16)(k1 * c - k2 * sn);
  K[base + 64] = (__bf16)(k2 * c + k1 * sn);
}

// V [token, nh*128+d]  ->  Vt [b, nh, d, s]
__global__ __launch_bounds__(256) void transpose_v_k(const __bf16* __restrict__ V,
                                                     __bf16* __restrict__ Vt) {
  int i = blockIdx.x * 256 + threadIdx.x;        // 2*32*128*2048 = 16.7M
  if (i >= B_ * NH_ * HD_ * S_) return;
  int s  = i & (S_ - 1);
  int d  = (i >> 11) & (HD_ - 1);
  int nh = (i >> 18) & (NH_ - 1);
  int b  = i >> 23;
  Vt[i] = V[(((size_t)(b * S_ + s)) << 12) + ((size_t)nh << 7) + d];
}

// ---------------------------------------------------------------- GEMM C = A * W^T
// A: [M,K] bf16 row-major. W: [N,K] bf16 row-major (Linear weight [out,in]).
// Block: 256 thr = 8 waves. Tile 128(M) x 128(N). W-tile double-buffered in LDS
// via global_load_async_to_lds_b128 (ASYNCcnt), layout [n][k] so each lane's
// B-fragment is one contiguous 32-byte LDS read.
__device__ __forceinline__ void gemm_step(const __bf16* __restrict__ cb,
                                          const __bf16* __restrict__ arow,
                                          int k0, int col, int hv, f32x8* acc) {
  bf16x8 alo = *(const bf16x8*)(arow + k0 + hv * 16);
  bf16x8 ahi = *(const bf16x8*)(arow + k0 + hv * 16 + 8);
  bf16x16 afrag = mk16(alo, ahi);
#pragma unroll
  for (int nt = 0; nt < 8; ++nt) {
    bf16x16 bfrag = *(const bf16x16*)(cb + (nt * 16 + col) * 32 + hv * 16);
    acc[nt] = __builtin_amdgcn_wmma_f32_16x16x32_bf16(
        false, afrag, false, bfrag, (short)0, acc[nt], false, false);
  }
}

__global__ __launch_bounds__(256) void gemm_bf16_nt(const __bf16* __restrict__ A,
                                                    const __bf16* __restrict__ W,
                                                    float* __restrict__ Cf,
                                                    __bf16* __restrict__ Cb,
                                                    int M, int N, int K) {
  __shared__ __bf16 ldsB[2][128 * 32];           // 2 x 8 KB double buffer
  const int tid  = threadIdx.x;
  const int wave = tid >> 5, lane = tid & 31;
  const int col  = lane & 15, hv = lane >> 4;
  const int bm = blockIdx.x * 128, bn = blockIdx.y * 128;
  const int row0 = bm + wave * 16;

  f32x8 acc[8];
#pragma unroll
  for (int nt = 0; nt < 8; ++nt) acc[nt] = vzero8();

  const __bf16* arow = A + (size_t)(row0 + col) * K;
  const int ln = tid >> 1;                       // 0..127 (W row within tile)
  const int lk = (tid & 1) * 16;                 // 0 or 16 (k chunk)
  const __bf16* wsrc = W + (size_t)(bn + ln) * K + lk;
  __bf16* lds0 = &ldsB[0][ln * 32 + lk];
  __bf16* lds1 = &ldsB[1][ln * 32 + lk];

  // prologue: async-fill tile 0 into buffer 0 (2 async ops per lane-slot)
  async_copy_b128(wsrc, lds0);
  async_copy_b128(wsrc + 8, lds0 + 8);

  const int KT = K / 32;
  // steady state: branch-free body, always prefetching the next tile
  for (int kt = 0; kt < KT - 1; ++kt) {
    const int k0 = kt * 32;
    __bf16* nb = (kt & 1) ? lds0 : lds1;          // tile kt+1 -> other buffer
    async_copy_b128(wsrc + k0 + 32, nb);
    async_copy_b128(wsrc + k0 + 40, nb + 8);
    wait_async_le2();                             // tile kt's 2 ops done (FIFO)
    __syncthreads();                              // all waves' portions visible

    __builtin_prefetch(arow + k0 + 64, 0, 1);     // global_prefetch_b8 on A stream

    gemm_step((kt & 1) ? ldsB[1] : ldsB[0], arow, k0, col, hv, acc);
    __syncthreads();                              // guard buffer reuse at kt+2
  }
  // drain: last tile, no further prefetch
  {
    const int k0 = (KT - 1) * 32;
    wait_async_le0();
    __syncthreads();
    gemm_step(((KT - 1) & 1) ? ldsB[1] : ldsB[0], arow, k0, col, hv, acc);
  }

#pragma unroll
  for (int nt = 0; nt < 8; ++nt)
#pragma unroll
    for (int j = 0; j < 8; ++j) {
      int r = row0 + j + 8 * hv;
      int c = bn + nt * 16 + col;
      if (Cf) Cf[(size_t)r * N + c] = acc[nt][j];
      else    Cb[(size_t)r * N + c] = (__bf16)acc[nt][j];
    }
}

// ---------------------------------------------------------------- flash attention
// grid: (S_/128, NH_, B_), 256 thr = 8 waves; wave owns 16 q-rows.
// Q,K: [token, nh*128+d] bf16 (RoPE applied). Vt: [b,nh,d,s] bf16. O: [token, nh*128+d].
__global__ __launch_bounds__(256) void flash_attn_k(const __bf16* __restrict__ Q,
                                                    const __bf16* __restrict__ K,
                                                    const __bf16* __restrict__ Vt,
                                                    __bf16* __restrict__ O) {
  __shared__ __bf16 ldsP[8 * 16 * 64];           // 16 KB, per-wave 16x64 P tile
  const int tid  = threadIdx.x;
  const int wave = tid >> 5, lane = tid & 31;
  const int col  = lane & 15, hv = lane >> 4;
  const int qtile = blockIdx.x * 128;
  const int nh = blockIdx.y, b = blockIdx.z;
  const int qrow0 = qtile + wave * 16;

  const __bf16* Qb = Q + ((size_t)(b * S_) << 12) + (nh << 7);
  const __bf16* Kb = K + ((size_t)(b * S_) << 12) + (nh << 7);
  const __bf16* Vb = Vt + (size_t)(b * NH_ + nh) * HD_ * S_;
  __bf16* ldsW = ldsP + wave * 16 * 64;

  float m_run[8], l_run[8];
  f32x8 oacc[8];
#pragma unroll
  for (int j = 0; j < 8; ++j) { m_run[j] = -1e30f; l_run[j] = 0.f; }
#pragma unroll
  for (int dt = 0; dt < 8; ++dt) oacc[dt] = vzero8();

  const float scale = 0.08838834764831845f;      // 1/sqrt(128)

  // preload Q A-fragments for the 4 K-steps over HD (reused every key block)
  bf16x16 qfrag[4];
  {
    const __bf16* qrow = Qb + (size_t)(qrow0 + col) * H_;
#pragma unroll
    for (int ks = 0; ks < 4; ++ks) {
      int d0 = ks * 32;
      bf16x8 lo = *(const bf16x8*)(qrow + d0 + hv * 16);
      bf16x8 hi = *(const bf16x8*)(qrow + d0 + hv * 16 + 8);
      qfrag[ks] = mk16(lo, hi);
    }
  }

  const int kend = qtile + 128;                  // causal: keys <= max q in tile
  for (int kb = 0; kb < kend; kb += 64) {
    // ---- S = Q * K^T  (16 x 64 scores per wave)
    f32x8 sacc[4];
#pragma unroll
    for (int nt = 0; nt < 4; ++nt) sacc[nt] = vzero8();
#pragma unroll
    for (int ks = 0; ks < 4; ++ks) {
      int d0 = ks * 32;
#pragma unroll
      for (int nt = 0; nt < 4; ++nt) {
        int key = kb + nt * 16 + col;
        bf16x16 bfrag = *(const bf16x16*)(Kb + (size_t)key * H_ + d0 + hv * 16);
        sacc[nt] = __builtin_amdgcn_wmma_f32_16x16x32_bf16(
            false, qfrag[ks], false, bfrag, (short)0, sacc[nt], false, false);
      }
    }

    // ---- online softmax (rows j + 8*hv live on 16-lane halves)
    float alpha[8];
#pragma unroll
    for (int j = 0; j < 8; ++j) {
      int q = qrow0 + j + 8 * hv;
      float sv[4];
      float rm = -1e30f;
#pragma unroll
      for (int nt = 0; nt < 4; ++nt) {
        int key = kb + nt * 16 + col;
        float s = sacc[nt][j] * scale;
        sv[nt] = (key <= q) ? s : -1e30f;
        rm = fmaxf(rm, sv[nt]);
      }
#pragma unroll
      for (int off = 1; off < 16; off <<= 1)
        rm = fmaxf(rm, __shfl_xor(rm, off, 32));
      float mn = fmaxf(m_run[j], rm);
      alpha[j] = __expf(m_run[j] - mn);
      m_run[j] = mn;
      float rs = 0.f;
#pragma unroll
      for (int nt = 0; nt < 4; ++nt) {
        float p = __expf(sv[nt] - mn);
        rs += p;
        ldsW[(j + 8 * hv) * 64 + nt * 16 + col] = (__bf16)p;
      }
#pragma unroll
      for (int off = 1; off < 16; off <<= 1)
        rs += __shfl_xor(rs, off, 32);
      l_run[j] = l_run[j] * alpha[j] + rs;
    }
#pragma unroll
    for (int dt = 0; dt < 8; ++dt)
#pragma unroll
      for (int j = 0; j < 8; ++j) oacc[dt][j] *= alpha[j];

    // ---- O += P * V   (P from per-wave LDS in A-layout; V from Vt, contiguous keys)
#pragma unroll
    for (int ks = 0; ks < 2; ++ks) {
      int k0 = ks * 32;
      const __bf16* prow = ldsW + col * 64 + k0;
      bf16x8 lo = *(const bf16x8*)(prow + hv * 16);
      bf16x8 hi = *(const bf16x8*)(prow + hv * 16 + 8);
      bf16x16 pfrag = mk16(lo, hi);
#pragma unroll
      for (int dt = 0; dt < 8; ++dt) {
        bf16x16 vfrag = *(const bf16x16*)(Vb + (size_t)(dt * 16 + col) * S_ + kb + k0 + hv * 16);
        oacc[dt] = __builtin_amdgcn_wmma_f32_16x16x32_bf16(
            false, pfrag, false, vfrag, (short)0, oacc[dt], false, false);
      }
    }
  }

  // ---- epilogue: O /= l, write bf16 [token, nh*128+d]
  __bf16* Ob = O + ((size_t)(b * S_) << 12) + (nh << 7);
#pragma unroll
  for (int dt = 0; dt < 8; ++dt)
#pragma unroll
    for (int j = 0; j < 8; ++j) {
      int q = qrow0 + j + 8 * hv;
      float v = oacc[dt][j] / l_run[j];
      Ob[(size_t)q * H_ + dt * 16 + col] = (__bf16)v;
    }
}

// ---------------------------------------------------------------- launch
extern "C" void kernel_launch(void* const* d_in, const int* in_sizes, int n_in,
                              void* d_out, int out_size, void* d_ws, size_t ws_size,
                              hipStream_t stream) {
  const float* hs = (const float*)d_in[0];
  // d_in[1] = attention_mask (causal; applied analytically)
  const float* Wq = (const float*)d_in[2];
  const float* Wk = (const float*)d_in[3];
  const float* Wv = (const float*)d_in[4];
  const float* Wo = (const float*)d_in[5];

  char* ws = (char*)d_ws;
  const size_t SZ = (size_t)H_ * (B_ * S_) * sizeof(__bf16);   // 32 MB
  __bf16* Xb  = (__bf16*)(ws + 0 * SZ);   // hidden bf16; reused as attn-out later
  __bf16* Wqb = (__bf16*)(ws + 1 * SZ);
  __bf16* Wkb = (__bf16*)(ws + 2 * SZ);
  __bf16* Wvb = (__bf16*)(ws + 3 * SZ);
  __bf16* Wob = (__bf16*)(ws + 4 * SZ);
  __bf16* Qb  = (__bf16*)(ws + 5 * SZ);
  __bf16* Kb  = (__bf16*)(ws + 6 * SZ);
  __bf16* Vb  = (__bf16*)(ws + 7 * SZ);
  __bf16* Vtb = (__bf16*)(ws + 8 * SZ);
  float*  cosT = (float*)(ws + 9 * SZ);
  float*  sinT = cosT + S_ * 64;

  const int nconv = H_ * H_;                     // 16.7M (== tokens*H too)
  const int cgrid = nconv / (256 * 4);
  f32_to_bf16_k<<<cgrid, 256, 0, stream>>>(hs, Xb, nconv);
  f32_to_bf16_k<<<cgrid, 256, 0, stream>>>(Wq, Wqb, nconv);
  f32_to_bf16_k<<<cgrid, 256, 0, stream>>>(Wk, Wkb, nconv);
  f32_to_bf16_k<<<cgrid, 256, 0, stream>>>(Wv, Wvb, nconv);
  f32_to_bf16_k<<<cgrid, 256, 0, stream>>>(Wo, Wob, nconv);

  rope_tables_k<<<(S_ * 64) / 256, 256, 0, stream>>>(cosT, sinT);

  dim3 ggrid((B_ * S_) / 128, H_ / 128);
  gemm_bf16_nt<<<ggrid, 256, 0, stream>>>(Xb, Wqb, nullptr, Qb, B_ * S_, H_, H_);
  gemm_bf16_nt<<<ggrid, 256, 0, stream>>>(Xb, Wkb, nullptr, Kb, B_ * S_, H_, H_);
  gemm_bf16_nt<<<ggrid, 256, 0, stream>>>(Xb, Wvb, nullptr, Vb, B_ * S_, H_, H_);

  rope_apply_k<<<((B_ * S_) * NH_ * 64) / 256, 256, 0, stream>>>(Qb, Kb, cosT, sinT);
  transpose_v_k<<<(B_ * NH_ * HD_ * S_) / 256, 256, 0, stream>>>(Vb, Vtb);

  flash_attn_k<<<dim3(S_ / 128, NH_, B_), 256, 0, stream>>>(Qb, Kb, Vtb, Xb);

  gemm_bf16_nt<<<ggrid, 256, 0, stream>>>(Xb, Wob, (float*)d_out, nullptr,
                                          B_ * S_, H_, H_);
}
